// DynamicAttentionNetwork_55413668053097
// MI455X (gfx1250) — compile-verified, matrix operand order checked by reading, not run on hardware
//
#include <hip/hip_runtime.h>
#include <hip/hip_bf16.h>
#include <math.h>

#define BDIM 4
#define NDIM 4096
#define HDIM 64
#define T1DIM 3072

typedef __attribute__((ext_vector_type(16))) __bf16 v16bf;
typedef __attribute__((ext_vector_type(8)))  float  v8f;

union BF16Frag {
  v16bf v;
  unsigned short u[16];
  unsigned int   d[8];
  uint4          q[2];
};

__device__ __forceinline__ unsigned short f2bf(float f) {
  unsigned int x = __float_as_uint(f);
  x += 0x7FFFu + ((x >> 16) & 1u);   // round-to-nearest-even
  return (unsigned short)(x >> 16);
}

// Pack two f32 -> one dword of two bf16 (lo in [15:0]) via v_perm_b32.
__device__ __forceinline__ unsigned int f2bf2(float lo, float hi) {
  unsigned int a = __float_as_uint(lo);
  a += 0x7FFFu + ((a >> 16) & 1u);
  unsigned int b = __float_as_uint(hi);
  b += 0x7FFFu + ((b >> 16) & 1u);
  // {S0,S1} = {b,a}; dst = {b[3],b[2],a[3],a[2]}
  return __builtin_amdgcn_perm(b, a, 0x07060302u);
}

__device__ __forceinline__ v8f wmma_bf16(v16bf a, v16bf b, v8f c) {
  // 8 args: (neg_a, A, neg_b, B, c_mod, C, reuse_a, reuse_b)
  return __builtin_amdgcn_wmma_f32_16x16x32_bf16(false, a, false, b, (short)0, c,
                                                 false, false);
}

// 16x32 bf16 fragment (A/B mirrored layout) from row-major bf16 data.
// lane l: row = l&15; K pattern {kb..kb+7, kb+16..kb+23}, kb = (l>>4)*8.
// Both 8-element chunks are contiguous 16B -> two b128 loads, zero VALU.
__device__ __forceinline__ v16bf frag_u16(const unsigned short* tile, int stride) {
  const int l = threadIdx.x & 31;
  const unsigned short* p = tile + (l & 15) * stride + ((l >> 4) << 3);
  BF16Frag f;
  f.q[0] = *(const uint4*)(p);
  f.q[1] = *(const uint4*)(p + 16);
  return f.v;
}

// Same fragment from a row-major f32 tile: float4 loads + v_perm packing.
__device__ __forceinline__ v16bf frag_f32(const float* tile, int stride) {
  const int l = threadIdx.x & 31;
  const float* p = tile + (l & 15) * stride + ((l >> 4) << 3);
  const float4 x0 = *(const float4*)(p);
  const float4 x1 = *(const float4*)(p + 4);
  const float4 y0 = *(const float4*)(p + 16);
  const float4 y1 = *(const float4*)(p + 20);
  BF16Frag f;
  f.d[0] = f2bf2(x0.x, x0.y);
  f.d[1] = f2bf2(x0.z, x0.w);
  f.d[2] = f2bf2(x1.x, x1.y);
  f.d[3] = f2bf2(x1.z, x1.w);
  f.d[4] = f2bf2(y0.x, y0.y);
  f.d[5] = f2bf2(y0.z, y0.w);
  f.d[6] = f2bf2(y1.x, y1.y);
  f.d[7] = f2bf2(y1.z, y1.w);
  return f.v;
}

// ---------------------------------------------------------------------------
// Kernel 1: Q = H*Wq^T, K = H*Wk^T (row-major bf16 [b][n][h]);
//           V = H*Wv^T stored transposed as bf16 [b][h][n] for the P*V stage.
// One wave per 16 rows of hidden_states.  One-shot code: keep loops rolled.
// ---------------------------------------------------------------------------
__global__ __launch_bounds__(128)
void qkv_kernel(const float* __restrict__ hidden,
                const float* __restrict__ Wq, const float* __restrict__ Wk,
                const float* __restrict__ Wv,
                unsigned short* __restrict__ Q, unsigned short* __restrict__ K,
                unsigned short* __restrict__ Vt) {
  const int wave = threadIdx.x >> 5;
  const int l    = threadIdx.x & 31;
  const int col  = l & 15;            // D-layout: N index
  const int rb   = (l >> 4) << 3;     // D-layout: M base (0 or 8)
  const int tile = blockIdx.x * 4 + wave;          // B*N/16 tiles total
  const int b    = tile / (NDIM / 16);
  const int n0   = (tile % (NDIM / 16)) * 16;

  const float* hrow = hidden + ((size_t)b * NDIM + n0) * HDIM;
  const v16bf a0 = frag_f32(hrow, HDIM);       // K = 0..31  (K-dim = H)
  const v16bf a1 = frag_f32(hrow + 32, HDIM);  // K = 32..63

  // Q projection
#pragma unroll 1
  for (int ot = 0; ot < 4; ++ot) {
    const float* wt = Wq + ot * 16 * HDIM;     // B[h][o] = W[o][h]
    v8f c = {};
    c = wmma_bf16(a0, frag_f32(wt, HDIM), c);
    c = wmma_bf16(a1, frag_f32(wt + 32, HDIM), c);
#pragma unroll
    for (int r = 0; r < 8; ++r)
      Q[((size_t)b * NDIM + n0 + rb + r) * HDIM + ot * 16 + col] = f2bf(c[r]);
  }
  // K projection
#pragma unroll 1
  for (int ot = 0; ot < 4; ++ot) {
    const float* wt = Wk + ot * 16 * HDIM;
    v8f c = {};
    c = wmma_bf16(a0, frag_f32(wt, HDIM), c);
    c = wmma_bf16(a1, frag_f32(wt + 32, HDIM), c);
#pragma unroll
    for (int r = 0; r < 8; ++r)
      K[((size_t)b * NDIM + n0 + rb + r) * HDIM + ot * 16 + col] = f2bf(c[r]);
  }
  // V projection, stored transposed [b][h][n]: rows rb..rb+7 contiguous -> b128.
#pragma unroll 1
  for (int ot = 0; ot < 4; ++ot) {
    const float* wt = Wv + ot * 16 * HDIM;
    v8f c = {};
    c = wmma_bf16(a0, frag_f32(wt, HDIM), c);
    c = wmma_bf16(a1, frag_f32(wt + 32, HDIM), c);
    uint4 pk;
    pk.x = f2bf2(c[0], c[1]);
    pk.y = f2bf2(c[2], c[3]);
    pk.z = f2bf2(c[4], c[5]);
    pk.w = f2bf2(c[6], c[7]);
    *(uint4*)(Vt + ((size_t)b * HDIM + ot * 16 + col) * NDIM + n0 + rb) = pk;
  }
}

// ---------------------------------------------------------------------------
// Kernel 2: fused masked attention (L2-row-norm) + branch update.
// One wave owns 16 query rows; block = 4 independent waves (64 rows).
// ---------------------------------------------------------------------------
__global__ __launch_bounds__(128)
void attn_update_kernel(const float* __restrict__ hidden,
                        const float* __restrict__ adj,
                        const unsigned short* __restrict__ Q,
                        const unsigned short* __restrict__ K,
                        const unsigned short* __restrict__ Vt,
                        const float* __restrict__ Wu_i, const float* __restrict__ bu_i,
                        const float* __restrict__ Wt_i, const float* __restrict__ bt_i,
                        const float* __restrict__ Wu_o, const float* __restrict__ bu_o,
                        const float* __restrict__ Wt_o, const float* __restrict__ bt_o,
                        const float* __restrict__ step_p,
                        float* __restrict__ out) {
  __shared__ unsigned short lds[4][16 * 64];   // per-wave staging buffer (2 KB each)
  const int wave = threadIdx.x >> 5;
  const int l    = threadIdx.x & 31;
  const int col  = l & 15;
  const int rb   = (l >> 4) << 3;
  const int b    = blockIdx.x / (NDIM / 64);
  const int n0   = (blockIdx.x % (NDIM / 64)) * 64 + wave * 16;

  const unsigned short* Qb = Q + ((size_t)b * NDIM + n0) * HDIM;
  const v16bf qa0 = frag_u16(Qb, HDIM);
  const v16bf qa1 = frag_u16(Qb + 32, HDIM);

  v8f acc0 = {}, acc1 = {}, acc2 = {}, acc3 = {};
  float ssq[8] = {0.f, 0.f, 0.f, 0.f, 0.f, 0.f, 0.f, 0.f};
  unsigned short* s_s = lds[wave];

  const unsigned short* Kbase = K + (size_t)b * NDIM * HDIM;
  const unsigned short* Vbase = Vt + (size_t)b * HDIM * NDIM;

#pragma unroll 2
  for (int m0 = 0; m0 < NDIM; m0 += 32) {
    // ---- S = Q * K^T for 32 keys (two 16x16 tiles) ----
    const unsigned short* Kb = Kbase + (size_t)m0 * HDIM;
    v8f s0 = {}, s1 = {};
    s0 = wmma_bf16(qa0, frag_u16(Kb, HDIM), s0);
    s0 = wmma_bf16(qa1, frag_u16(Kb + 32, HDIM), s0);
    s1 = wmma_bf16(qa0, frag_u16(Kb + 16 * HDIM, HDIM), s1);
    s1 = wmma_bf16(qa1, frag_u16(Kb + 16 * HDIM + 32, HDIM), s1);

    // ---- mask with adjacency[m][n], accumulate sum of squares, stage bf16 S ----
    // D-layout element (M=i, N=j): i = rb+r, j = col.  mask = adj[m0+j][n0+i].
    const float* am0 = adj + (size_t)(m0 + col) * NDIM + n0 + rb;
    const float* am1 = am0 + (size_t)16 * NDIM;

    // Pull the next chunk's streaming operands toward the WGP.
    if (m0 + 32 < NDIM) {
      __builtin_prefetch(Kb + 32 * HDIM, 0, 3);
      __builtin_prefetch(am0 + (size_t)32 * NDIM, 0, 3);
    }

#pragma unroll
    for (int r = 0; r < 8; ++r) {
      const float v0 = s0[r] * am0[r];
      const float v1 = s1[r] * am1[r];
      ssq[r] += v0 * v0 + v1 * v1;
      s_s[(rb + r) * 32 + col]      = f2bf(v0);
      s_s[(rb + r) * 32 + 16 + col] = f2bf(v1);
    }

    // ---- acc += S * V  (A = S from LDS, B = Vt rows are V columns) ----
    const v16bf pa = frag_u16(s_s, 32);   // compiler inserts s_wait_dscnt
    const unsigned short* Vb = Vbase + m0;
    acc0 = wmma_bf16(pa, frag_u16(Vb + (size_t)0  * NDIM, NDIM), acc0);
    acc1 = wmma_bf16(pa, frag_u16(Vb + (size_t)16 * NDIM, NDIM), acc1);
    acc2 = wmma_bf16(pa, frag_u16(Vb + (size_t)32 * NDIM, NDIM), acc2);
    acc3 = wmma_bf16(pa, frag_u16(Vb + (size_t)48 * NDIM, NDIM), acc3);
  }

  // ---- per-row L2 normalization: reduce ssq over each 16-lane half ----
#pragma unroll
  for (int r = 0; r < 8; ++r) {
    float t = ssq[r];
    t += __shfl_xor(t, 1, 32);
    t += __shfl_xor(t, 2, 32);
    t += __shfl_xor(t, 4, 32);
    t += __shfl_xor(t, 8, 32);        // masks <16 stay inside each half-wave
    const float rn = (t > 0.f) ? rsqrtf(t) : 0.f;
    acc0[r] *= rn; acc1[r] *= rn; acc2[r] *= rn; acc3[r] *= rn;
  }

  // ---- stage acc as bf16 16x64 in LDS for the update GEMM A-operand ----
  unsigned short* s_a = lds[wave];
#pragma unroll
  for (int r = 0; r < 8; ++r) {
    s_a[(rb + r) * 64 + col]      = f2bf(acc0[r]);
    s_a[(rb + r) * 64 + 16 + col] = f2bf(acc1[r]);
    s_a[(rb + r) * 64 + 32 + col] = f2bf(acc2[r]);
    s_a[(rb + r) * 64 + 48 + col] = f2bf(acc3[r]);
  }
  const v16bf aa0 = frag_u16(s_a, 64);
  const v16bf aa1 = frag_u16(s_a + 32, 64);

  const float* hrow = hidden + ((size_t)b * NDIM + n0) * HDIM;
  const v16bf ha0 = frag_f32(hrow, HDIM);
  const v16bf ha1 = frag_f32(hrow + 32, HDIM);

  const bool inter = (n0 < T1DIM);     // wave-uniform (T1 % 64 == 0)
  const float* Wu = inter ? Wu_i : Wu_o;
  const float* bu = inter ? bu_i : bu_o;
  const float* Wt = inter ? Wt_i : Wt_o;
  const float* bt = inter ? bt_i : bt_o;
  const float step = step_p[0];

  // One-shot epilogue: keep rolled to minimize code footprint.
#pragma unroll 1
  for (int ot = 0; ot < 4; ++ot) {
    const float* wut = Wu + ot * 16 * HDIM;
    v8f u = {};
    u = wmma_bf16(aa0, frag_f32(wut, HDIM), u);
    u = wmma_bf16(aa1, frag_f32(wut + 32, HDIM), u);
    const float* wtt = Wt + ot * 16 * HDIM;
    v8f tv = {};
    tv = wmma_bf16(ha0, frag_f32(wtt, HDIM), tv);
    tv = wmma_bf16(ha1, frag_f32(wtt + 32, HDIM), tv);

    const int o = ot * 16 + col;
    const float buv = bu[o];
    const float btv = bt[o];
#pragma unroll
    for (int r = 0; r < 8; ++r) {
      const size_t idx = ((size_t)b * NDIM + n0 + rb + r) * HDIM + o;
      const float x  = tv[r] + btv;
      const float dt = 2.f / (1.f + __expf(-x)) - 1.f;   // 2*sigmoid - 1
      out[idx] = hidden[idx] + step * dt * (u[r] + buv);
    }
  }
}

extern "C" void kernel_launch(void* const* d_in, const int* in_sizes, int n_in,
                              void* d_out, int out_size, void* d_ws, size_t ws_size,
                              hipStream_t stream) {
  (void)in_sizes; (void)n_in; (void)out_size; (void)ws_size;
  const float* hidden = (const float*)d_in[0];
  // d_in[1] (input_states) is unused by the reference.
  const float* adj    = (const float*)d_in[2];
  const float* Wq     = (const float*)d_in[3];
  const float* Wk     = (const float*)d_in[4];
  const float* Wv     = (const float*)d_in[5];
  const float* Wu_i   = (const float*)d_in[6];
  const float* bu_i   = (const float*)d_in[7];
  const float* Wt_i   = (const float*)d_in[8];
  const float* bt_i   = (const float*)d_in[9];
  const float* Wu_o   = (const float*)d_in[10];
  const float* bu_o   = (const float*)d_in[11];
  const float* Wt_o   = (const float*)d_in[12];
  const float* bt_o   = (const float*)d_in[13];
  const float* step_p = (const float*)d_in[14];
  float* out = (float*)d_out;

  unsigned short* Q  = (unsigned short*)d_ws;                 // 2 MB
  unsigned short* K  = Q + (size_t)BDIM * NDIM * HDIM;        // 2 MB
  unsigned short* Vt = K + (size_t)BDIM * NDIM * HDIM;        // 2 MB, [b][h][n]

  dim3 blk(128);   // 4 wave32s
  qkv_kernel<<<BDIM * NDIM / 64, blk, 0, stream>>>(hidden, Wq, Wk, Wv, Q, K, Vt);
  attn_update_kernel<<<BDIM * (NDIM / 64), blk, 0, stream>>>(
      hidden, adj, Q, K, Vt, Wu_i, bu_i, Wt_i, bt_i, Wu_o, bu_o, Wt_o, bt_o,
      step_p, out);
}
